// LSTM_893353197792
// MI455X (gfx1250) — matrix operand order repaired and checked by reading, not compiled, not run on hardware
//
#include <hip/hip_runtime.h>

// Problem dims
#define B_   256
#define T_   128
#define I_   512
#define H_   1024
#define O_   512
#define G4   (4 * H_)        // 4096 gate columns
#define NT_G (G4 / 16)       // 256 N-tiles for gates
#define NT_H (H_ / 16)       // 64 j-tiles within hidden dim
#define KT_I (I_ / 32)       // 16 K-tiles over input dim
#define KT_H (H_ / 32)       // 32 K-tiles over hidden dim
#define KT_ALL (KT_I + KT_H) // 48 combined K-steps per timestep
#define MT_B (B_ / 16)       // 16 M-tiles over batch
#define NT_O (O_ / 16)       // 32 N-tiles for FC output

typedef __attribute__((ext_vector_type(16))) __bf16 v16bf;
typedef __attribute__((ext_vector_type(8)))  float  v8f;

#define WMMA_BF16(a, b, c) \
    __builtin_amdgcn_wmma_f32_16x16x32_bf16(false, (a), false, (b), (short)0, (c), false, false)

// ---- helpers -------------------------------------------------------------

__device__ __forceinline__ unsigned short f32_to_bf16(float f) {
    union { float f; unsigned u; } v; v.f = f;
    unsigned u = v.u;
    if ((u & 0x7F800000u) == 0x7F800000u) return (unsigned short)(u >> 16); // inf/nan: truncate
    unsigned r = u + 0x7FFFu + ((u >> 16) & 1u);                            // RNE
    return (unsigned short)(r >> 16);
}

// A-matrix 16x32 bf16 lane layout (ISA 7.12.2):
//   lane = half*16 + row ; k = (e<8) ? half*8+e : 16+half*8+(e-8)
// inverse: half = (kk>>3)&1 ; e = (kk&7) + ((kk>>4)<<3)
__device__ __forceinline__ int a_lane(int row, int kk) { return (((kk >> 3) & 1) << 4) + row; }
__device__ __forceinline__ int a_elem(int kk)          { return (kk & 7) + ((kk >> 4) << 3); }

// B-matrix 32x16 bf16 lane layout: lane = half*16 + col ; k = (e<8) ? half*8+e : 16+half*8+(e-8)
__device__ __forceinline__ int b_lane(int col, int kk) { return (((kk >> 3) & 1) << 4) + col; }
__device__ __forceinline__ int b_elem(int kk)          { return (kk & 7) + ((kk >> 4) << 3); }

__device__ __forceinline__ float sigmoidf_(float x) { return 1.f / (1.f + __expf(-x)); }

// ---- one-time packing kernels -------------------------------------------

// W_ih [4096,512] row-major f32 -> B-frags [KT_I][NT_G][32 lanes][16]
__global__ void pack_wih(const float* __restrict__ w, unsigned short* __restrict__ frag) {
    int idx = blockIdx.x * blockDim.x + threadIdx.x;   // 4096*512
    int g = idx >> 9, i = idx & 511;
    int kt = i >> 5, kk = i & 31, tn = g >> 4, col = g & 15;
    frag[(((kt * NT_G + tn) << 9)) + b_lane(col, kk) * 16 + b_elem(kk)] = f32_to_bf16(w[idx]);
}

// W_hh [4096,1024] -> B-frags [KT_H][NT_G][32][16]
__global__ void pack_whh(const float* __restrict__ w, unsigned short* __restrict__ frag) {
    int idx = blockIdx.x * blockDim.x + threadIdx.x;   // 4096*1024
    int g = idx >> 10, j = idx & 1023;
    int kt = j >> 5, kk = j & 31, tn = g >> 4, col = g & 15;
    frag[(((kt * NT_G + tn) << 9)) + b_lane(col, kk) * 16 + b_elem(kk)] = f32_to_bf16(w[idx]);
}

// W_fc [512,1024] -> B-frags [KT_H][NT_O][32][16]
__global__ void pack_wfc(const float* __restrict__ w, unsigned short* __restrict__ frag) {
    int idx = blockIdx.x * blockDim.x + threadIdx.x;   // 512*1024
    int o = idx >> 10, j = idx & 1023;
    int kt = j >> 5, kk = j & 31, tn = o >> 4, col = o & 15;
    frag[(((kt * NT_O + tn) << 9)) + b_lane(col, kk) * 16 + b_elem(kk)] = f32_to_bf16(w[idx]);
}

// zero c (f32) and h bf16 fragment slot 0 (h0 = c0 = 0)
__global__ void init_state(float* __restrict__ c, unsigned short* __restrict__ hfrag0) {
    int idx = blockIdx.x * blockDim.x + threadIdx.x;   // 262144
    c[idx] = 0.f; hfrag0[idx] = 0;
}

// x[:,t,:] f32 -> A-frags [MT_B][KT_I][32][16] into a double-buffer slot
__global__ void pack_x(const float* __restrict__ x, int t, unsigned short* __restrict__ frag) {
    int idx = blockIdx.x * blockDim.x + threadIdx.x;   // 256*512
    int b = idx >> 9, i = idx & 511;
    int tm = b >> 4, row = b & 15, kt = i >> 5, kk = i & 31;
    frag[(((tm * KT_I + kt) << 9)) + a_lane(row, kk) * 16 + a_elem(kk)] =
        f32_to_bf16(x[b * (T_ * I_) + t * I_ + i]);
}

// ---- fully fused LSTM timestep ------------------------------------------
// One wave owns the four gate tiles (i,f,g,o) of one (batch-tile tm, j-tile tj).
// Grid is only 1024 waves (recurrence-limited) => ~1 wave/SIMD, so occupancy is
// useless; amdgpu_waves_per_eu(1) lifts the register ceiling so the scheduler
// can keep many b128 loads in flight across the fully unrolled 48-step K loop.
__global__ void __launch_bounds__(256)
__attribute__((amdgpu_waves_per_eu(1)))
lstm_step(const unsigned short* __restrict__ xfrag_cur,
          const unsigned short* __restrict__ hfrag_cur,
          unsigned short* __restrict__ hfrag_nxt,
          const unsigned short* __restrict__ wihfrag,
          const unsigned short* __restrict__ whhfrag,
          const float* __restrict__ b_ih,
          const float* __restrict__ b_hh,
          float* __restrict__ cst,
          float* __restrict__ hst,          // written on last step only
          const float* __restrict__ x,      // to pack x[t+1]
          unsigned short* __restrict__ xfrag_nxt,
          int t) {
    int lane = threadIdx.x & 31;
    int wave = threadIdx.x >> 5;
    int tile = blockIdx.x * 8 + wave;           // 1024 tiles exactly (128 blocks x 8 waves)
    int tm = tile >> 6;                         // batch tile  0..15
    int tj = tile & (NT_H - 1);                 // j tile      0..63
    int lo = lane * 16;

    // ---- hoist all epilogue-side loads: their latency hides under the GEMM
    int n  = tj * 16 + (lane & 15);             // hidden index j, 0..1023
    int m0 = tm * 16 + ((lane >> 4) << 3);
    float bi = b_ih[n]          + b_hh[n];
    float bf = b_ih[H_ + n]     + b_hh[H_ + n];
    float bg = b_ih[2 * H_ + n] + b_hh[2 * H_ + n];
    float bo = b_ih[3 * H_ + n] + b_hh[3 * H_ + n];
    float cold[8];
#pragma unroll
    for (int v = 0; v < 8; ++v) cold[v] = cst[(m0 + v) * H_ + n];
    float xv[4];
    int g = blockIdx.x * blockDim.x + threadIdx.x;   // 0..32767 (for x[t+1] pack)
    bool more = (t + 1 < T_);
    if (more) {
#pragma unroll
        for (int r = 0; r < 4; ++r) {
            int eidx = r * 32768 + g;
            xv[r] = x[(eidx >> 9) * (T_ * I_) + (t + 1) * I_ + (eidx & 511)];
        }
    }

    v8f ai = {}, af = {}, ag = {}, ao = {};

    // fragment base for combined K-step ks (0..47): x-phase then h-phase
    auto a_base = [&](int ks) -> const unsigned short* {
        return (ks < KT_I) ? (xfrag_cur + ((tm * KT_I + ks) << 9))
                           : (hfrag_cur + ((tm * KT_H + (ks - KT_I)) << 9));
    };
    auto b_base = [&](int ks) -> const unsigned short* {
        return (ks < KT_I) ? (wihfrag + ((ks * NT_G + tj) << 9))
                           : (whhfrag + (((ks - KT_I) * NT_G + tj) << 9));
    };

    // prologue: fragments for ks = 0
    const unsigned short* ab = a_base(0);
    const unsigned short* bb = b_base(0);
    v16bf a  = *(const v16bf*)(ab + lo);
    v16bf b0 = *(const v16bf*)(bb + lo);
    v16bf b1 = *(const v16bf*)(bb + (64  << 9) + lo);
    v16bf b2 = *(const v16bf*)(bb + (128 << 9) + lo);
    v16bf b3 = *(const v16bf*)(bb + (192 << 9) + lo);

#pragma unroll
    for (int ks = 0; ks < KT_ALL - 1; ++ks) {
        // prefetch ks+1 into fresh registers (loads overlap the WMMAs below)
        const unsigned short* abn = a_base(ks + 1);
        const unsigned short* bbn = b_base(ks + 1);
        v16bf an  = *(const v16bf*)(abn + lo);
        v16bf bn0 = *(const v16bf*)(bbn + lo);
        v16bf bn1 = *(const v16bf*)(bbn + (64  << 9) + lo);
        v16bf bn2 = *(const v16bf*)(bbn + (128 << 9) + lo);
        v16bf bn3 = *(const v16bf*)(bbn + (192 << 9) + lo);
        // compute ks (operands already resident)
        ai = WMMA_BF16(a, b0, ai);
        af = WMMA_BF16(a, b1, af);
        ag = WMMA_BF16(a, b2, ag);
        ao = WMMA_BF16(a, b3, ao);
        a = an; b0 = bn0; b1 = bn1; b2 = bn2; b3 = bn3;
    }
    // epilogue K-step
    ai = WMMA_BF16(a, b0, ai);
    af = WMMA_BF16(a, b1, af);
    ag = WMMA_BF16(a, b2, ag);
    ao = WMMA_BF16(a, b3, ao);

    // ---- in-register LSTM cell update (pure VALU + stores) ---------------
    // C/D layout: lane<16 -> M=v, lane>=16 -> M=8+v ; N = lane&15
    int kt_h = n >> 5, kk = n & 31;
    unsigned short* hbase = hfrag_nxt + ((tm * KT_H + kt_h) << 9) + a_elem(kk);
    int fl = a_lane(0, kk);                     // + row gives the fragment lane
    bool last = (t == T_ - 1);
#pragma unroll
    for (int v = 0; v < 8; ++v) {
        int m = m0 + v;
        int idx = m * H_ + n;
        float cn = sigmoidf_(af[v] + bf) * cold[v] + sigmoidf_(ai[v] + bi) * tanhf(ag[v] + bg);
        float hn = sigmoidf_(ao[v] + bo) * tanhf(cn);
        cst[idx] = cn;
        if (last) hst[idx] = hn;
        hbase[(fl + (m & 15)) * 16] = f32_to_bf16(hn);
    }

    // ---- pack x[t+1] into the other slot (values preloaded above) --------
    if (more) {
#pragma unroll
        for (int r = 0; r < 4; ++r) {
            int eidx = r * 32768 + g;                   // 131072 elements total
            int b = eidx >> 9, i = eidx & 511;
            int ptm = b >> 4, prow = b & 15, pkt = i >> 5, pkk = i & 31;
            xfrag_nxt[((ptm * KT_I + pkt) << 9) + a_lane(prow, pkk) * 16 + a_elem(pkk)] =
                f32_to_bf16(xv[r]);
        }
    }
}

// ---- final FC: out = h*W_fc^T + b_fc ------------------------------------

__global__ void fc_gemm(const unsigned short* __restrict__ hfrag,
                        const unsigned short* __restrict__ wfcfrag,
                        const float* __restrict__ b_fc,
                        float* __restrict__ out) {               // [256][512]
    int lane = threadIdx.x & 31;
    int wave = threadIdx.x >> 5;
    int tile = blockIdx.x * 8 + wave;                // 512 tiles exactly
    int tm = tile >> 5;                              // / NT_O
    int tn = tile & (NT_O - 1);
    int lo = lane * 16;
    v8f acc = {};
#pragma unroll 4
    for (int kt = 0; kt < KT_H; ++kt) {
        v16bf a = *(const v16bf*)(hfrag   + ((tm * KT_H + kt) << 9) + lo);
        v16bf b = *(const v16bf*)(wfcfrag + ((kt * NT_O + tn) << 9) + lo);
        acc = WMMA_BF16(a, b, acc);
    }
    int n  = tn * 16 + (lane & 15);
    float bias = b_fc[n];
    int m0 = tm * 16 + ((lane >> 4) << 3);
    float* op = out + m0 * O_ + n;
#pragma unroll
    for (int v = 0; v < 8; ++v) op[v * O_] = acc[v] + bias;
}

// ---- write h, c to output (reference returns out, then h[None], c[None]) -

__global__ void copy_hc(const float* __restrict__ h, const float* __restrict__ c,
                        float* __restrict__ out) {
    int idx = blockIdx.x * blockDim.x + threadIdx.x;   // 256*1024
    out[B_ * O_ + idx]           = h[idx];
    out[B_ * O_ + B_ * H_ + idx] = c[idx];
}

// ---- host-side launch ----------------------------------------------------

extern "C" void kernel_launch(void* const* d_in, const int* in_sizes, int n_in,
                              void* d_out, int out_size, void* d_ws, size_t ws_size,
                              hipStream_t stream) {
    (void)in_sizes; (void)n_in; (void)out_size; (void)ws_size;
    const float* x    = (const float*)d_in[0];
    const float* W_ih = (const float*)d_in[1];
    const float* W_hh = (const float*)d_in[2];
    const float* b_ih = (const float*)d_in[3];
    const float* b_hh = (const float*)d_in[4];
    const float* W_fc = (const float*)d_in[5];
    const float* b_fc = (const float*)d_in[6];
    float* out = (float*)d_out;

    // workspace carve-up (~16.5 MB total)
    char* p = (char*)d_ws;
    unsigned short* wihfrag = (unsigned short*)p; p += (size_t)KT_I * NT_G * 512 * 2;     // 4 MB
    unsigned short* whhfrag = (unsigned short*)p; p += (size_t)KT_H * NT_G * 512 * 2;     // 8 MB
    unsigned short* wfcfrag = (unsigned short*)p; p += (size_t)KT_H * NT_O * 512 * 2;     // 1 MB
    unsigned short* xfrag   = (unsigned short*)p; p += (size_t)2 * MT_B * KT_I * 512 * 2; // 0.5 MB (2 slots)
    unsigned short* hfrag   = (unsigned short*)p; p += (size_t)2 * MT_B * KT_H * 512 * 2; // 1 MB (2 slots)
    float* cst = (float*)p; p += (size_t)B_ * H_ * 4;                                     // 1 MB
    float* hst = (float*)p; p += (size_t)B_ * H_ * 4;                                     // 1 MB
    const int XSLOT = MT_B * KT_I * 512;   // ushorts per x slot
    const int HSLOT = MT_B * KT_H * 512;   // ushorts per h slot

    // one-time packing / init
    pack_wih  <<<(4096 * 512)  / 256, 256, 0, stream>>>(W_ih, wihfrag);
    pack_whh  <<<(4096 * 1024) / 256, 256, 0, stream>>>(W_hh, whhfrag);
    pack_wfc  <<<(512 * 1024)  / 256, 256, 0, stream>>>(W_fc, wfcfrag);
    init_state<<<(B_ * H_) / 256, 256, 0, stream>>>(cst, hfrag);   // zeros h-slot 0
    pack_x    <<<(B_ * I_) / 256, 256, 0, stream>>>(x, 0, xfrag);  // x[0] -> slot 0

    // one fused kernel per timestep: GEMM + cell update + repack, double-buffered
    for (int t = 0; t < T_; ++t) {
        const unsigned short* xs = xfrag + (t & 1) * XSLOT;
        unsigned short*       xn = xfrag + ((t + 1) & 1) * XSLOT;
        const unsigned short* hs = hfrag + (t & 1) * HSLOT;
        unsigned short*       hn = hfrag + ((t + 1) & 1) * HSLOT;
        lstm_step<<<(MT_B * NT_H) / 8, 256, 0, stream>>>(xs, hs, hn, wihfrag, whhfrag,
                                                         b_ih, b_hh, cst, hst, x, xn, t);
    }

    // final FC on last hidden state (t=127 wrote h-frags into slot 0) + emit h_n, c_n
    fc_gemm<<<(MT_B * NT_O) / 8, 256, 0, stream>>>(hfrag, wfcfrag, b_fc, out);
    copy_hc<<<(B_ * H_) / 256, 256, 0, stream>>>(hst, cst, out);
}
